// Attention_74594991997623
// MI455X (gfx1250) — matrix operand order repaired and checked by reading, not compiled, not run on hardware
//
#include <hip/hip_runtime.h>
#include <stdint.h>

#define SQd 2048
#define SKd 2048
#define NB  4
#define DD  512

typedef __attribute__((ext_vector_type(16))) __bf16 v16bf;
typedef __attribute__((ext_vector_type(8)))  __bf16 v8bf;
typedef __attribute__((ext_vector_type(4)))  __bf16 v4bf;
typedef __attribute__((ext_vector_type(8)))  float  v8f;

#define SCALE 0.04419417382415922f   /* 1/sqrt(512) */
#define NEGBIG (-1.0e30f)

#if defined(__has_builtin)
#  if __has_builtin(__builtin_amdgcn_sched_barrier)
#    define SCHED_FENCE() __builtin_amdgcn_sched_barrier(0)
#  endif
#endif
#ifndef SCHED_FENCE
#  define SCHED_FENCE() ((void)0)
#endif

__device__ __forceinline__ __bf16 f2bf(float x) { return (__bf16)x; }

__device__ __forceinline__ void load8(const float* __restrict__ p, float* d) {
  const float4 x = *(const float4*)p;
  const float4 y = *(const float4*)(p + 4);
  d[0] = x.x; d[1] = x.y; d[2] = x.z; d[3] = x.w;
  d[4] = y.x; d[5] = y.y; d[6] = y.z; d[7] = y.w;
}

__device__ __forceinline__ v16bf pack16(const float* a, const float* b) {
  v16bf r;
#pragma unroll
  for (int i = 0; i < 8; ++i) { r[i] = f2bf(a[i]); r[i + 8] = f2bf(b[i]); }
  return r;
}

// ============================ pre-pass kernels ============================

// K (f32, [k][b][d]) -> Kb (bf16, same layout). 8 elements per thread.
__global__ __launch_bounds__(256)
void convK_kernel(const float* __restrict__ K, __bf16* __restrict__ Kb) {
  const size_t i = ((size_t)blockIdx.x * 256 + threadIdx.x) * 8;
  const float4 a = *(const float4*)(K + i);
  const float4 c = *(const float4*)(K + i + 4);
  v8bf r;
  r[0] = f2bf(a.x); r[1] = f2bf(a.y); r[2] = f2bf(a.z); r[3] = f2bf(a.w);
  r[4] = f2bf(c.x); r[5] = f2bf(c.y); r[6] = f2bf(c.z); r[7] = f2bf(c.w);
  *(v8bf*)(Kb + i) = r;
}

// V (f32, [k][b][d]) -> Vt (bf16, [b][d][k]) via LDS-tiled transpose.
__global__ __launch_bounds__(256)
void transV_kernel(const float* __restrict__ V, __bf16* __restrict__ Vt) {
  __shared__ __bf16 tile[32][33];
  const int kb = blockIdx.x * 32;
  const int db = blockIdx.y * 32;
  const int b  = blockIdx.z;
  const int tf = threadIdx.x & 31;   // fast (coalesced) index
  const int ts = threadIdx.x >> 5;   // 0..7
#pragma unroll
  for (int p = 0; p < 4; ++p) {
    const int k = ts + p * 8;
    tile[k][tf] = f2bf(V[(size_t)(kb + k) * (NB * DD) + (size_t)b * DD + db + tf]);
  }
  __syncthreads();
#pragma unroll
  for (int p = 0; p < 4; ++p) {
    const int d = ts + p * 8;
    Vt[(size_t)b * DD * SKd + (size_t)(db + d) * SKd + kb + tf] = tile[tf][d];
  }
}

// ===================== main kernel: pre-converted bf16 =====================
// Block = 128 threads = 4 waves; 16 queries; wave w owns D slice [128w,128w+128).
// B-fragments are contiguous 32B bf16 global loads; a sched_barrier pins the
// load group above the WMMA burst so all loads issue before the first wait.
__global__ __launch_bounds__(128)
void fa_bf16_kernel(const float* __restrict__ Q, const __bf16* __restrict__ Kb,
                    const __bf16* __restrict__ Vt, const uint8_t* __restrict__ kmask,
                    float* __restrict__ Out)
{
  __shared__ float  sbuf[16 * 32];
  __shared__ __bf16 pbuf[16 * 32];
  __shared__ float  mrow[16], lrow[16], arow[16];

  const int tid    = threadIdx.x;
  const int lane   = tid & 31;
  const int wave   = tid >> 5;
  const int hi     = lane >> 4;
  const int ln     = lane & 15;
  const int qbase  = blockIdx.x * 16;
  const int b      = blockIdx.y;
  const int dslice = wave * 128;

  for (int i = tid; i < 16 * 32; i += 128) sbuf[i] = 0.0f;
  if (tid < 16) { mrow[tid] = NEGBIG; lrow[tid] = 0.0f; arow[tid] = 1.0f; }

  // Q A-fragments for this wave's D slice (converted once, kept in registers)
  const float* qrow = Q + (size_t)(qbase + ln) * (NB * DD) + (size_t)b * DD;
  v16bf qa[4];
#pragma unroll
  for (int c = 0; c < 4; ++c) {
    float lo8[8], hi8[8];
    const int base = dslice + c * 32 + (hi ? 16 : 0);
    load8(qrow + base,      lo8);
    load8(qrow + base + 8, hi8);
    qa[c] = pack16(lo8, hi8);
  }

  v8f o[8] = {};
  const __bf16* vbase = Vt + (size_t)b * DD * SKd;

  __syncthreads();   // sbuf/stats initialized before first atomics

  const int nsteps = (qbase + 15) / 32 + 1;
  for (int kt = 0; kt < nsteps; ++kt) {
    const int kbase = kt * 32;

    // ---- QK^T partial over this wave's D slice.
    // Stage all 8 B-fragments (16 b128 loads) and pin them above the WMMAs.
    v16bf kf[8];
    {
      const __bf16* kr0 = Kb + ((size_t)(kbase + ln)      * NB + b) * DD + dslice + (hi ? 16 : 0);
      const __bf16* kr1 = Kb + ((size_t)(kbase + 16 + ln) * NB + b) * DD + dslice + (hi ? 16 : 0);
#pragma unroll
      for (int c = 0; c < 4; ++c) {
        kf[2 * c]     = *(const v16bf*)(kr0 + c * 32);
        kf[2 * c + 1] = *(const v16bf*)(kr1 + c * 32);
      }
    }
    SCHED_FENCE();
    v8f s0 = {}, s1 = {};
#pragma unroll
    for (int c = 0; c < 4; ++c) {
      s0 = __builtin_amdgcn_wmma_f32_16x16x32_bf16(false, qa[c], false, kf[2 * c],     (short)0, s0, false, false);
      s1 = __builtin_amdgcn_wmma_f32_16x16x32_bf16(false, qa[c], false, kf[2 * c + 1], (short)0, s1, false, false);
    }
#pragma unroll
    for (int j = 0; j < 8; ++j) {
      const int r = j + 8 * hi;
      atomicAdd(&sbuf[r * 32 + ln],      s0[j]);
      atomicAdd(&sbuf[r * 32 + 16 + ln], s1[j]);
    }
    __syncthreads();

    // ---- online softmax, one thread per query row
    if (tid < 16) {
      const int r  = tid;
      const int qi = qbase + r;
      float sv[32];
#pragma unroll
      for (int c2 = 0; c2 < 32; ++c2) {
        const int kj = kbase + c2;
        float x = sbuf[r * 32 + c2] * SCALE;
        if (kj > qi || kmask[(size_t)kj * NB + b]) x = NEGBIG;
        sv[c2] = x;
        sbuf[r * 32 + c2] = 0.0f;
      }
      const float mold = mrow[r];
      float mx = mold;
#pragma unroll
      for (int c2 = 0; c2 < 32; ++c2) mx = fmaxf(mx, sv[c2]);
      const float alpha = __expf(mold - mx);
      float sum = 0.0f;
#pragma unroll
      for (int c2 = 0; c2 < 32; ++c2) {
        const float p = (sv[c2] <= 0.5f * NEGBIG) ? 0.0f : __expf(sv[c2] - mx);
        pbuf[r * 32 + c2] = f2bf(p);
        sum += p;
      }
      lrow[r] = lrow[r] * alpha + sum;
      mrow[r] = mx;
      arow[r] = alpha;
    }
    __syncthreads();

    // ---- rescale O, then PV with direct global V fragments
    float av[8];
#pragma unroll
    for (int j = 0; j < 8; ++j) av[j] = arow[j + 8 * hi];
#pragma unroll
    for (int t = 0; t < 8; ++t)
#pragma unroll
      for (int j = 0; j < 8; ++j) o[t][j] *= av[j];

    {
      const __bf16* pr = &pbuf[ln * 32 + (hi ? 16 : 0)];
      v8bf plo = *(const v8bf*)pr;
      v8bf phi = *(const v8bf*)(pr + 8);
      v16bf pa;
#pragma unroll
      for (int i = 0; i < 8; ++i) { pa[i] = plo[i]; pa[i + 8] = phi[i]; }

      v16bf vf[8];
#pragma unroll
      for (int t = 0; t < 8; ++t)
        vf[t] = *(const v16bf*)(vbase + (size_t)(dslice + t * 16 + ln) * SKd
                                      + kbase + (hi ? 16 : 0));
#pragma unroll
      for (int t = 0; t < 8; ++t)
        o[t] = __builtin_amdgcn_wmma_f32_16x16x32_bf16(false, pa, false, vf[t], (short)0, o[t], false, false);
    }
  }

  float invl[8];
#pragma unroll
  for (int j = 0; j < 8; ++j) {
    const float l = lrow[j + 8 * hi];
    invl[j] = (l > 0.0f) ? 1.0f / l : 0.0f;
  }
#pragma unroll
  for (int t = 0; t < 8; ++t) {
#pragma unroll
    for (int j = 0; j < 8; ++j) {
      const int row = j + 8 * hi;
      Out[((size_t)(qbase + row) * NB + b) * DD + dslice + t * 16 + ln] = o[t][j] * invl[j];
    }
  }
}

// ============== fallback (self-contained, no workspace needed) ==============
__global__ __launch_bounds__(128)
void fa_fallback_kernel(const float* __restrict__ Q, const float* __restrict__ K,
                        const float* __restrict__ V, const uint8_t* __restrict__ kmask,
                        float* __restrict__ Out)
{
  __shared__ __bf16 vbuf[DD * 32];
  __shared__ float  sbuf[16 * 32];
  __shared__ __bf16 pbuf[16 * 32];
  __shared__ float  mrow[16], lrow[16], arow[16];

  const int tid    = threadIdx.x;
  const int lane   = tid & 31;
  const int wave   = tid >> 5;
  const int hi     = lane >> 4;
  const int ln     = lane & 15;
  const int qbase  = blockIdx.x * 16;
  const int b      = blockIdx.y;
  const int dslice = wave * 128;

  for (int i = tid; i < 16 * 32; i += 128) sbuf[i] = 0.0f;
  if (tid < 16) { mrow[tid] = NEGBIG; lrow[tid] = 0.0f; arow[tid] = 1.0f; }

  const float* qrow = Q + (size_t)(qbase + ln) * (NB * DD) + (size_t)b * DD;
  v16bf qa[4];
#pragma unroll
  for (int c = 0; c < 4; ++c) {
    float lo8[8], hi8[8];
    const int base = dslice + c * 32 + (hi ? 16 : 0);
    load8(qrow + base,      lo8);
    load8(qrow + base + 8, hi8);
    qa[c] = pack16(lo8, hi8);
  }

  v8f o[8] = {};
  const int nsteps = (qbase + 15) / 32 + 1;
  for (int kt = 0; kt < nsteps; ++kt) {
    const int kbase = kt * 32;
    __syncthreads();
    {
      const int d0 = tid * 4;
#pragma unroll
      for (int k = 0; k < 32; k += 4) {
        float4 f[4];
#pragma unroll
        for (int u = 0; u < 4; ++u)
          f[u] = *(const float4*)(V + (size_t)(kbase + k + u) * (NB * DD)
                                    + (size_t)b * DD + d0);
#pragma unroll
        for (int j = 0; j < 4; ++j) {
          v4bf pk;
#pragma unroll
          for (int u = 0; u < 4; ++u)
            pk[u] = f2bf(((const float*)&f[u])[j]);
          *(v4bf*)&vbuf[(d0 + j) * 32 + k] = pk;
        }
      }
    }
    __syncthreads();

    v8f s0 = {}, s1 = {};
#pragma unroll
    for (int c = 0; c < 4; ++c) {
      const int dk = dslice + c * 32;
      const float* kr0 = K + (size_t)(kbase + ln)      * (NB * DD) + (size_t)b * DD + dk + (hi ? 16 : 0);
      const float* kr1 = K + (size_t)(kbase + 16 + ln) * (NB * DD) + (size_t)b * DD + dk + (hi ? 16 : 0);
      float x0[8], x1[8];
      load8(kr0, x0); load8(kr0 + 8, x1);
      v16bf kb0 = pack16(x0, x1);
      load8(kr1, x0); load8(kr1 + 8, x1);
      v16bf kb1 = pack16(x0, x1);
      s0 = __builtin_amdgcn_wmma_f32_16x16x32_bf16(false, qa[c], false, kb0, (short)0, s0, false, false);
      s1 = __builtin_amdgcn_wmma_f32_16x16x32_bf16(false, qa[c], false, kb1, (short)0, s1, false, false);
    }
#pragma unroll
    for (int j = 0; j < 8; ++j) {
      const int r = j + 8 * hi;
      atomicAdd(&sbuf[r * 32 + ln],      s0[j]);
      atomicAdd(&sbuf[r * 32 + 16 + ln], s1[j]);
    }
    __syncthreads();

    if (tid < 16) {
      const int r  = tid;
      const int qi = qbase + r;
      float sv[32];
#pragma unroll
      for (int c2 = 0; c2 < 32; ++c2) {
        const int kj = kbase + c2;
        float x = sbuf[r * 32 + c2] * SCALE;
        if (kj > qi || kmask[(size_t)kj * NB + b]) x = NEGBIG;
        sv[c2] = x;
        sbuf[r * 32 + c2] = 0.0f;
      }
      const float mold = mrow[r];
      float mx = mold;
#pragma unroll
      for (int c2 = 0; c2 < 32; ++c2) mx = fmaxf(mx, sv[c2]);
      const float alpha = __expf(mold - mx);
      float sum = 0.0f;
#pragma unroll
      for (int c2 = 0; c2 < 32; ++c2) {
        const float p = (sv[c2] <= 0.5f * NEGBIG) ? 0.0f : __expf(sv[c2] - mx);
        pbuf[r * 32 + c2] = f2bf(p);
        sum += p;
      }
      lrow[r] = lrow[r] * alpha + sum;
      mrow[r] = mx;
      arow[r] = alpha;
    }
    __syncthreads();

    float av[8];
#pragma unroll
    for (int j = 0; j < 8; ++j) av[j] = arow[j + 8 * hi];
#pragma unroll
    for (int t = 0; t < 8; ++t)
#pragma unroll
      for (int j = 0; j < 8; ++j) o[t][j] *= av[j];

    {
      const __bf16* pr = &pbuf[ln * 32 + (hi ? 16 : 0)];
      v8bf plo = *(const v8bf*)pr;
      v8bf phi = *(const v8bf*)(pr + 8);
      v16bf pa;
#pragma unroll
      for (int i = 0; i < 8; ++i) { pa[i] = plo[i]; pa[i + 8] = phi[i]; }
#pragma unroll
      for (int t = 0; t < 8; ++t) {
        const __bf16* vr = &vbuf[(dslice + t * 16 + ln) * 32 + (hi ? 16 : 0)];
        v8bf vlo = *(const v8bf*)vr;
        v8bf vhi = *(const v8bf*)(vr + 8);
        v16bf vb;
#pragma unroll
        for (int i = 0; i < 8; ++i) { vb[i] = vlo[i]; vb[i + 8] = vhi[i]; }
        o[t] = __builtin_amdgcn_wmma_f32_16x16x32_bf16(false, pa, false, vb, (short)0, o[t], false, false);
      }
    }
  }

  float invl[8];
#pragma unroll
  for (int j = 0; j < 8; ++j) {
    const float l = lrow[j + 8 * hi];
    invl[j] = (l > 0.0f) ? 1.0f / l : 0.0f;
  }
#pragma unroll
  for (int t = 0; t < 8; ++t) {
#pragma unroll
    for (int j = 0; j < 8; ++j) {
      const int row = j + 8 * hi;
      Out[((size_t)(qbase + row) * NB + b) * DD + dslice + t * 16 + ln] = o[t][j] * invl[j];
    }
  }
}

// ================================ launcher ================================
extern "C" void kernel_launch(void* const* d_in, const int* in_sizes, int n_in,
                              void* d_out, int out_size, void* d_ws, size_t ws_size,
                              hipStream_t stream) {
  (void)in_sizes; (void)n_in; (void)out_size;
  const float*   Q    = (const float*)d_in[0];
  const float*   K    = (const float*)d_in[1];
  const float*   V    = (const float*)d_in[2];
  const uint8_t* mask = (const uint8_t*)d_in[3];
  float* Out = (float*)d_out;

  const size_t kbBytes = (size_t)SKd * NB * DD * sizeof(__bf16);  // 8 MB
  const size_t vtBytes = (size_t)NB * DD * SKd * sizeof(__bf16);  // 8 MB
  dim3 grid(SQd / 16, NB);

  if (ws_size >= kbBytes + vtBytes) {
    __bf16* Kb = (__bf16*)d_ws;
    __bf16* Vt = (__bf16*)((char*)d_ws + kbBytes);
    convK_kernel<<<(SKd * NB * DD) / (256 * 8), 256, 0, stream>>>(K, Kb);
    transV_kernel<<<dim3(SKd / 32, DD / 32, NB), 256, 0, stream>>>(V, Vt);
    fa_bf16_kernel<<<grid, 128, 0, stream>>>(Q, Kb, Vt, mask, Out);
  } else {
    fa_fallback_kernel<<<grid, 128, 0, stream>>>(Q, K, V, mask, Out);
  }
}
